// Attention_11046655885816
// MI455X (gfx1250) — matrix operand order repaired and checked by reading, not compiled, not run on hardware
//
#include <hip/hip_runtime.h>

// ---------------------------------------------------------------------------
// Attention: out = softmax(QWq (KWk)^T - keymask*1e12) (VWv), masked by Q_len
// B=4, L=2048, D=1024, H=16, DH=64.  f16 WMMA, f32 accumulation.
//
// Both GEMM stages are computed TRANSPOSED so every WMMA fragment is a
// contiguous vector load and flash-softmax state is lane-local:
//   proj:  out^T = W^T * X^T     (A=W^T contiguous, B=X^T contiguous)
//   attn:  S^T   = K * Q^T       (lane owns a query column)
//          O^T  += V^T * P^T     (P^T B-fragment is already in-lane; V^T
//                                 A-fragment is two 16B loads per lane)
// v3: 2x wider tiles per wave (K/V/W fragments feed two WMMA sets each),
//     all loads issued ahead of the consuming WMMAs.
// ---------------------------------------------------------------------------

typedef __attribute__((ext_vector_type(16))) _Float16 v16h;
typedef __attribute__((ext_vector_type(8)))  _Float16 v8h;
typedef __attribute__((ext_vector_type(4)))  _Float16 v4h;
typedef __attribute__((ext_vector_type(8)))  float    v8f;

static constexpr int CB  = 4;
static constexpr int CL  = 2048;
static constexpr int CD  = 1024;
static constexpr int CH  = 16;
static constexpr int CDH = 64;
static constexpr int CE  = CH * CDH;   // 1024

#define WMMA_F16(a, b, c) \
    __builtin_amdgcn_wmma_f32_16x16x32_f16(false, (a), false, (b), (short)0, (c), false, false)

// ---------------------------------------------------------------------------
// Prep 1: convert X (Q_seq/K_seq/V_seq) fp32 -> f16, straight copy.
// ---------------------------------------------------------------------------
__global__ __launch_bounds__(256) void cvt_kernel(
    const float* __restrict__ X0, const float* __restrict__ X1,
    const float* __restrict__ X2,
    _Float16* __restrict__ Y0, _Float16* __restrict__ Y1,
    _Float16* __restrict__ Y2)
{
    const int z = blockIdx.y;
    const float* X = (z == 0) ? X0 : (z == 1) ? X1 : X2;
    _Float16*    Y = (z == 0) ? Y0 : (z == 1) ? Y1 : Y2;

    const size_t i = ((size_t)blockIdx.x * 256 + threadIdx.x) * 8;
    const float4* p = reinterpret_cast<const float4*>(X + i);
    const float4 a = p[0], b = p[1];
    v8h h;
    h[0] = (_Float16)a.x; h[1] = (_Float16)a.y;
    h[2] = (_Float16)a.z; h[3] = (_Float16)a.w;
    h[4] = (_Float16)b.x; h[5] = (_Float16)b.y;
    h[6] = (_Float16)b.z; h[7] = (_Float16)b.w;
    *reinterpret_cast<v8h*>(Y + i) = h;
}

// ---------------------------------------------------------------------------
// Prep 2: W[d][e] fp32 -> WT[e][d] f16 via 32x32 LDS tile transpose.
// ---------------------------------------------------------------------------
__global__ __launch_bounds__(256) void trw_kernel(
    const float* __restrict__ W0, const float* __restrict__ W1,
    const float* __restrict__ W2,
    _Float16* __restrict__ T0, _Float16* __restrict__ T1,
    _Float16* __restrict__ T2)
{
    const int z = blockIdx.y;
    const float* W = (z == 0) ? W0 : (z == 1) ? W1 : W2;
    _Float16*    T = (z == 0) ? T0 : (z == 1) ? T1 : T2;

    const int tile = blockIdx.x;                 // 1024 tiles of 32x32
    const int r0 = (tile >> 5) * 32;             // d
    const int c0 = (tile & 31) * 32;             // e

    __shared__ __align__(16) _Float16 tl[32][36];

    const int r = threadIdx.x >> 3;              // 0..31
    const int c = (threadIdx.x & 7) * 4;         // 0,4,..,28
    const float4 v =
        *reinterpret_cast<const float4*>(W + (size_t)(r0 + r) * CE + c0 + c);
    tl[r][c + 0] = (_Float16)v.x; tl[r][c + 1] = (_Float16)v.y;
    tl[r][c + 2] = (_Float16)v.z; tl[r][c + 3] = (_Float16)v.w;
    __syncthreads();

    v4h o;
    o[0] = tl[c + 0][r]; o[1] = tl[c + 1][r];
    o[2] = tl[c + 2][r]; o[3] = tl[c + 3][r];
    *reinterpret_cast<v4h*>(T + (size_t)(c0 + r) * CD + r0 + c) = o;
}

// ---------------------------------------------------------------------------
// Kernel 1: projection GEMM, transposed: out^T = W^T * X^T.
// 8 waves/block; wave tile = 64 features x 32 tokens (A fragments reused by
// two token groups). No LDS, no barriers.
//   z==0: Q -> Qh [B,H,L,DH]   z==1: K -> Kh [B,H,L,DH]   z==2: V -> Vt [B,H,DH,L]
// ---------------------------------------------------------------------------
__global__ __launch_bounds__(256) void proj_kernel(
    const _Float16* __restrict__ Xq, const _Float16* __restrict__ Xk,
    const _Float16* __restrict__ Xv,
    const _Float16* __restrict__ WTq, const _Float16* __restrict__ WTk,
    const _Float16* __restrict__ WTv,
    _Float16* __restrict__ Qh, _Float16* __restrict__ Kh,
    _Float16* __restrict__ Vt)
{
    const int z = blockIdx.z;
    const _Float16* X  = (z == 0) ? Xq  : (z == 1) ? Xk  : Xv;
    const _Float16* WT = (z == 0) ? WTq : (z == 1) ? WTk : WTv;

    const int lane = threadIdx.x & 31;
    const int wave = threadIdx.x >> 5;
    const int lm   = lane & 15;
    const int lh   = lane >> 4;

    const int tok0 = blockIdx.x * 256 + wave * 32;  // token columns (2 groups)
    const int fe0  = blockIdx.y * 64;               // feature rows

    const _Float16* xrow0 = X + (size_t)(tok0 + lm) * CD;
    const _Float16* xrow1 = X + (size_t)(tok0 + 16 + lm) * CD;
    const _Float16* wrow  = WT + (size_t)(fe0 + lm) * CD;

    v8f acc[2][4] = {};

    for (int k = 0; k < CD; k += 32) {
        // ---- issue ALL loads for this k-step first ----
        v16h aw[4];
        #pragma unroll
        for (int n = 0; n < 4; ++n) {
            const _Float16* wr = wrow + (size_t)(n * 16) * CD;
            const v8h c0 = *reinterpret_cast<const v8h*>(wr + k + lh * 16);
            const v8h c1 = *reinterpret_cast<const v8h*>(wr + k + lh * 16 + 8);
            #pragma unroll
            for (int e = 0; e < 8; ++e) { aw[n][e] = c0[e]; aw[n][8 + e] = c1[e]; }
        }
        const v16h bx0 = *reinterpret_cast<const v16h*>(xrow0 + k + lh * 16);
        const v16h bx1 = *reinterpret_cast<const v16h*>(xrow1 + k + lh * 16);

        // ---- 8 WMMAs reusing the 4 A fragments ----
        #pragma unroll
        for (int n = 0; n < 4; ++n) {
            acc[0][n] = WMMA_F16(aw[n], bx0, acc[0][n]);
            acc[1][n] = WMMA_F16(aw[n], bx1, acc[1][n]);
        }
    }

    // C^T layout: VGPR i = feature row (i + 8*lh), lane col = token lm
    #pragma unroll
    for (int g = 0; g < 2; ++g) {
        const int r = tok0 + g * 16 + lm;        // flat token
        const int b = r >> 11;
        const int l = r & 2047;
        #pragma unroll
        for (int n = 0; n < 4; ++n) {
            const int e0  = fe0 + n * 16 + lh * 8;  // first feature of this run
            const int h   = e0 >> 6;
            const int dh0 = e0 & 63;
            v8h hv;
            #pragma unroll
            for (int i = 0; i < 8; ++i) hv[i] = (_Float16)acc[g][n][i];
            if (z == 2) {
                #pragma unroll
                for (int i = 0; i < 8; ++i)
                    Vt[(((size_t)b * CH + h) * CDH + dh0 + i) * CL + l] = hv[i];
            } else {
                _Float16* dst = (z == 0) ? Qh : Kh;
                *reinterpret_cast<v8h*>(
                    &dst[(((size_t)b * CH + h) * CL + l) * CDH + dh0]) = hv;
            }
        }
    }
}

// ---------------------------------------------------------------------------
// Kernel 2: flash attention (transposed). One wave = 32 query columns
// (two groups); K/V fragments loaded once per 32-key block feed 16 WMMAs.
// Lane-local online softmax, one cross-half shfl_xor per reduction. No LDS.
// ---------------------------------------------------------------------------
__global__ __launch_bounds__(256) void attn_kernel(
    const _Float16* __restrict__ Qh, const _Float16* __restrict__ Kh,
    const _Float16* __restrict__ Vt,
    const int* __restrict__ Q_len, const int* __restrict__ V_len,
    float* __restrict__ Out)
{
    const int lane = threadIdx.x & 31;
    const int wave = threadIdx.x >> 5;
    const int lm   = lane & 15;
    const int lh   = lane >> 4;

    const int b  = blockIdx.z;
    const int h  = blockIdx.y;
    const int q0 = blockIdx.x * 256 + wave * 32;

    const int qlen = Q_len[b];
    const int vlen = V_len[b];

    const _Float16* Qbh = Qh + ((size_t)b * CH + h) * CL * CDH;
    const _Float16* Kbh = Kh + ((size_t)b * CH + h) * CL * CDH;
    const _Float16* Vbh = Vt + ((size_t)b * CH + h) * CDH * CL;

    // B = Q^T fragments (loop invariant): group g -> query q0+g*16+lm;
    // element f -> dh = s*32 + lh*16 + f  (contiguous 32B loads)
    v16h bq[2][2];
    #pragma unroll
    for (int g = 0; g < 2; ++g) {
        const _Float16* qrow = Qbh + (size_t)(q0 + g * 16 + lm) * CDH;
        bq[g][0] = *reinterpret_cast<const v16h*>(qrow + lh * 16);
        bq[g][1] = *reinterpret_cast<const v16h*>(qrow + 32 + lh * 16);
    }

    v8f   o[2][4] = {};
    float m[2]    = {-3.0e38f, -3.0e38f};
    float lsum[2] = {0.0f, 0.0f};

    for (int kb = 0; kb < CL; kb += 32) {
        if (kb + 32 < CL) {
            __builtin_prefetch(Kbh + (size_t)(kb + 32 + lm) * CDH, 0, 0);
            __builtin_prefetch(Vbh + (size_t)lm * CL + kb + 32, 0, 0);
        }

        // ---- issue ALL K fragment loads (A = K, rows kb+j*16+lm) ----
        v16h ak[2][2];                         // [s = dh step][j = key frag]
        #pragma unroll
        for (int s = 0; s < 2; ++s) {
            const _Float16* k0 = Kbh + (size_t)(kb + lm) * CDH + s * 32;
            const _Float16* k1 = Kbh + (size_t)(kb + 16 + lm) * CDH + s * 32;
            const v8h a0l = *reinterpret_cast<const v8h*>(k0 + lh * 16);
            const v8h a0h = *reinterpret_cast<const v8h*>(k0 + lh * 16 + 8);
            const v8h a1l = *reinterpret_cast<const v8h*>(k1 + lh * 16);
            const v8h a1h = *reinterpret_cast<const v8h*>(k1 + lh * 16 + 8);
            #pragma unroll
            for (int e = 0; e < 8; ++e) {
                ak[s][0][e] = a0l[e]; ak[s][0][8 + e] = a0h[e];
                ak[s][1][e] = a1l[e]; ak[s][1][8 + e] = a1h[e];
            }
        }
        // ---- issue V fragment loads early (overlap with S + softmax) ----
        v16h av[4];
        #pragma unroll
        for (int n = 0; n < 4; ++n) {
            const _Float16* vr = Vbh + (size_t)(n * 16 + lm) * CL + kb;
            const v8h v0 = *reinterpret_cast<const v8h*>(vr + lh * 8);
            const v8h v1 = *reinterpret_cast<const v8h*>(vr + 16 + lh * 8);
            #pragma unroll
            for (int e = 0; e < 8; ++e) { av[n][e] = v0[e]; av[n][8 + e] = v1[e]; }
        }

        // ---- S^T = K Q^T : 8 WMMAs over 2 query groups ----
        v8f sg[2][2] = {};                     // [g][j]
        #pragma unroll
        for (int s = 0; s < 2; ++s)
            #pragma unroll
            for (int g = 0; g < 2; ++g) {
                sg[g][0] = WMMA_F16(ak[s][0], bq[g][s], sg[g][0]);
                sg[g][1] = WMMA_F16(ak[s][1], bq[g][s], sg[g][1]);
            }

        // ---- per-group mask + lane-local online softmax ----
        const int kbase = kb + 8 * lh;
        v16h pp[2];
        #pragma unroll
        for (int g = 0; g < 2; ++g) {
            #pragma unroll
            for (int i = 0; i < 8; ++i) {
                sg[g][0][i] += (kbase + i      < vlen) ? 0.0f : -1.0e12f;
                sg[g][1][i] += (kbase + 16 + i < vlen) ? 0.0f : -1.0e12f;
            }
            float cur = sg[g][0][0];
            #pragma unroll
            for (int i = 1; i < 8; ++i) cur = fmaxf(cur, sg[g][0][i]);
            #pragma unroll
            for (int i = 0; i < 8; ++i) cur = fmaxf(cur, sg[g][1][i]);
            cur = fmaxf(cur, __shfl_xor(cur, 16, 32));
            const float mn    = fmaxf(m[g], cur);
            const float alpha = __expf(m[g] - mn);

            // p values land exactly in the B-fragment slots of P^T
            float ps = 0.0f;
            #pragma unroll
            for (int i = 0; i < 8; ++i) {
                const float p = __expf(sg[g][0][i] - mn);
                ps += p; pp[g][i] = (_Float16)p;
            }
            #pragma unroll
            for (int i = 0; i < 8; ++i) {
                const float p = __expf(sg[g][1][i] - mn);
                ps += p; pp[g][8 + i] = (_Float16)p;
            }
            ps += __shfl_xor(ps, 16, 32);
            lsum[g] = lsum[g] * alpha + ps;
            m[g]    = mn;
            #pragma unroll
            for (int n = 0; n < 4; ++n)
                #pragma unroll
                for (int i = 0; i < 8; ++i) o[g][n][i] *= alpha;
        }

        // ---- O^T += V^T P^T : 8 WMMAs reusing the 4 V fragments ----
        #pragma unroll
        for (int n = 0; n < 4; ++n) {
            o[0][n] = WMMA_F16(av[n], pp[0], o[0][n]);
            o[1][n] = WMMA_F16(av[n], pp[1], o[1][n]);
        }
    }

    // ---- normalize + query mask; O^T VGPR i = dh row n*16 + 8*lh + i ----
    #pragma unroll
    for (int g = 0; g < 2; ++g) {
        const int   q     = q0 + g * 16 + lm;
        const float inv   = (lsum[g] > 0.0f) ? (1.0f / lsum[g]) : 0.0f;
        const float scale = (q < qlen) ? inv : 0.0f;
        float* orow = Out + ((size_t)b * CL + q) * CE + h * CDH;
        #pragma unroll
        for (int n = 0; n < 4; ++n) {
            v8f r;
            #pragma unroll
            for (int i = 0; i < 8; ++i) r[i] = o[g][n][i] * scale;
            *reinterpret_cast<v8f*>(orow + n * 16 + lh * 8) = r;
        }
    }
}

// ---------------------------------------------------------------------------
extern "C" void kernel_launch(void* const* d_in, const int* in_sizes, int n_in,
                              void* d_out, int out_size, void* d_ws, size_t ws_size,
                              hipStream_t stream)
{
    const float* Q_seq = (const float*)d_in[0];
    const float* K_seq = (const float*)d_in[1];
    const float* V_seq = (const float*)d_in[2];
    const int*   Q_len = (const int*)d_in[3];
    const int*   V_len = (const int*)d_in[4];
    const float* WQ    = (const float*)d_in[5];
    const float* WK    = (const float*)d_in[6];
    const float* WV    = (const float*)d_in[7];

    const size_t tens = (size_t)CB * CL * CD;    // 8.4M elements
    const size_t wsz  = (size_t)CD * CE;         // 1M elements

    _Float16* Qh  = (_Float16*)d_ws;             // [B,H,L,DH]
    _Float16* Kh  = Qh  + tens;
    _Float16* Vt  = Kh  + tens;                  // [B,H,DH,L]
    _Float16* Xq  = Vt  + tens;                  // f16 copies of inputs
    _Float16* Xk  = Xq  + tens;
    _Float16* Xv  = Xk  + tens;
    _Float16* WTq = Xv  + tens;                  // W transposed, f16
    _Float16* WTk = WTq + wsz;
    _Float16* WTv = WTk + wsz;

    dim3 cg((unsigned)(tens / 2048), 3);         // 8 elems/thread
    cvt_kernel<<<cg, 256, 0, stream>>>(Q_seq, K_seq, V_seq, Xq, Xk, Xv);

    dim3 tg(1024, 3);                            // 32x32 tiles
    trw_kernel<<<tg, 256, 0, stream>>>(WQ, WK, WV, WTq, WTk, WTv);

    dim3 pg(32, 16, 3);                          // 256-token x 64-feature x QKV
    proj_kernel<<<pg, 256, 0, stream>>>(Xq, Xk, Xv, WTq, WTk, WTv, Qh, Kh, Vt);

    dim3 ag(8, 16, 4);                           // 256-query tiles x H x B
    attn_kernel<<<ag, 256, 0, stream>>>(Qh, Kh, Vt, Q_len, V_len, (float*)d_out);
}
